// QuantumLayer_78700980732389
// MI455X (gfx1250) — matrix operand order repaired and checked by reading, not compiled, not run on hardware
//
#include <hip/hip_runtime.h>

#define NQ      10
#define DIM     1024
#define TM      32          // batch rows per workgroup
#define THREADS 256

typedef __attribute__((ext_vector_type(16))) __bf16       v16bf;
typedef __attribute__((ext_vector_type(8)))  float        v8f;
typedef __attribute__((ext_vector_type(4)))  unsigned int v4u;

// ---- bf16 helpers (round-to-nearest-even), kept as ushort for storage ----
__device__ __forceinline__ unsigned short f2bf(float v) {
  union { float f; unsigned u; } a; a.f = v;
  unsigned u = a.u;
  unsigned r = (u + 0x7FFFu + ((u >> 16) & 1u)) >> 16;
  return (unsigned short)r;
}
__device__ __forceinline__ float bf2f(unsigned short b) {
  union { unsigned u; float f; } a; a.u = ((unsigned)b) << 16; return a.f;
}

// ---- CDNA5 async global->LDS copy of 32 contiguous bytes (2 x B128) ----
// INST_OFFSET is added to BOTH the LDS and global addresses (ISA 15.18.3),
// so one base pair + offset:16 covers the full 32B slot. Tracked by ASYNCcnt.
__device__ __forceinline__ void async_copy_32B(unsigned lds_byte_addr,
                                               const void* gsrc) {
  const unsigned long long ga = (unsigned long long)(uintptr_t)gsrc;
  asm volatile("global_load_async_to_lds_b128 %0, %1, off"
               :: "v"(lds_byte_addr), "v"(ga) : "memory");
  asm volatile("global_load_async_to_lds_b128 %0, %1, off offset:16"
               :: "v"(lds_byte_addr), "v"(ga) : "memory");
}

// =====================================================================
// Kernel 1: build B = U^T (K-major), split into bf16 hi/lo.
// One workgroup per basis column k: simulate the 40-gate circuit on e_k
// in LDS (1024 complex amplitudes = 8KB), then scatter the column into
// Bt rows: Re -> row n', Im -> row 1024+n'.  Bt layout: [2048][1024] bf16.
// =====================================================================
__global__ __launch_bounds__(THREADS) void qsel_build_bt(
    const float* __restrict__ wts,
    unsigned short* __restrict__ bthi,
    unsigned short* __restrict__ btlo)
{
  __shared__ float sr[DIM];
  __shared__ float si[DIM];
  const int k   = blockIdx.x;
  const int tid = threadIdx.x;

  for (int i = tid; i < DIM; i += THREADS) { sr[i] = (i == k) ? 1.f : 0.f; si[i] = 0.f; }
  __syncthreads();

  for (int l = 0; l < 2; ++l) {
    // ---- Rot on every wire ----
    for (int q = 0; q < NQ; ++q) {
      const float phi = wts[(l * NQ + q) * 3 + 0];
      const float th  = wts[(l * NQ + q) * 3 + 1];
      const float om  = wts[(l * NQ + q) * 3 + 2];
      float c, s, sa, ca, sb, cb;
      __sincosf(0.5f * th, &s, &c);
      __sincosf(0.5f * (phi + om), &sa, &ca);
      __sincosf(0.5f * (phi - om), &sb, &cb);
      // U00=(ca,-sa)c  U01=(-cb,-sb)s  U10=(cb,-sb)s  U11=(ca,sa)c
      const float u00r =  ca * c, u00i = -sa * c;
      const float u01r = -cb * s, u01i = -sb * s;
      const float u10r =  cb * s, u10i = -sb * s;
      const float u11r =  ca * c, u11i =  sa * c;
      const int m = 1 << (9 - q);
      for (int p = tid; p < 512; p += THREADS) {
        const int i0 = ((p & ~(m - 1)) << 1) | (p & (m - 1));
        const int i1 = i0 | m;
        const float a0r = sr[i0], a0i = si[i0], a1r = sr[i1], a1i = si[i1];
        sr[i0] = u00r * a0r - u00i * a0i + u01r * a1r - u01i * a1i;
        si[i0] = u00r * a0i + u00i * a0r + u01r * a1i + u01i * a1r;
        sr[i1] = u10r * a0r - u10i * a0i + u11r * a1r - u11i * a1i;
        si[i1] = u10r * a0i + u10i * a0r + u11r * a1i + u11i * a1r;
      }
      __syncthreads();
    }
    // ---- ring of CNOTs, range r ----
    const int r = (l % (NQ - 1)) + 1;
    for (int q = 0; q < NQ; ++q) {
      const int pc = 9 - q;
      const int pt = 9 - ((q + r) % NQ);
      const int lo = pc < pt ? pc : pt;
      const int hi = pc < pt ? pt : pc;
      const int mlo = 1 << lo, mhi = 1 << hi;
      if (tid < 256) {
        int v = tid;
        v = ((v & ~(mlo - 1)) << 1) | (v & (mlo - 1));
        v = ((v & ~(mhi - 1)) << 1) | (v & (mhi - 1));
        const int i0 = v | (1 << pc);      // control=1, target=0
        const int i1 = i0 | (1 << pt);
        const float tr = sr[i0]; sr[i0] = sr[i1]; sr[i1] = tr;
        const float ti = si[i0]; si[i0] = si[i1]; si[i1] = ti;
      }
      __syncthreads();
    }
  }

  // ---- write column k as Bt rows (K-major => GEMM B frags are contiguous) ----
  for (int n = tid; n < DIM; n += THREADS) {
    const float vr = sr[n], vi = si[n];
    const unsigned short hr = f2bf(vr), hi2 = f2bf(vi);
    bthi[(size_t)n * DIM + k]         = hr;
    btlo[(size_t)n * DIM + k]         = f2bf(vr - bf2f(hr));
    bthi[(size_t)(DIM + n) * DIM + k] = hi2;
    btlo[(size_t)(DIM + n) * DIM + k] = f2bf(vi - bf2f(hi2));
  }
}

// =====================================================================
// Kernel 2: Y = x * Bt^T via bf16x3-split WMMA, fused PauliZ epilogue.
//   out[b][q] = (P - 2*Z_q)/P  with P = sum_k |Y|^2, Z_q = bit-masked sum
// 8 waves: mt = wave&1 (M-tile of 16 rows), sub = wave>>1 (N-subtile).
// B chunks are double-buffered via GLOBAL_LOAD_ASYNC_TO_LDS (ASYNCcnt).
// =====================================================================
__global__ __launch_bounds__(THREADS) void qsel_gemm(
    const float* __restrict__ x,
    const unsigned short* __restrict__ bthi,
    const unsigned short* __restrict__ btlo,
    float* __restrict__ out)
{
  // LDS: A-frags hi/lo 2*64KB, B ping/pong 2*16KB (p-buffer aliases) = 160KB
  __shared__ unsigned short lds[81920];
  unsigned short* aHi  = lds;                   // [2 mt][32 kci][32 lane][16]
  unsigned short* aLo  = lds + 32768;
  float*          pbuf = (float*)(lds + 65536); // aliased [32 rows][64 cols]

  const int tid = threadIdx.x;
  const int t   = tid & 31;
  const int w   = tid >> 5;
  const int mt  = w & 1;
  const int sub = w >> 1;
  const int m0  = blockIdx.x * TM;

  const unsigned ldsBase = (unsigned)(size_t)(lds);
  const unsigned bByte0  = ldsBase + 131072u;   // B buffer 0 byte address

  // ---- stage A rows m0..m0+31 into WMMA A-fragment layout (bf16 hi/lo) ----
  for (int o = tid; o < TM * 128; o += THREADS) {
    const int row = o >> 7;
    const int k   = (o & 127) << 3;
    const float4* xv = (const float4*)(x + (size_t)(m0 + row) * DIM + k);
    const float4 f0 = xv[0], f1 = xv[1];
    const float vals[8] = {f0.x, f0.y, f0.z, f0.w, f1.x, f1.y, f1.z, f1.w};
    union { unsigned short h[8]; v4u u; } H, L;
#pragma unroll
    for (int i = 0; i < 8; ++i) {
      const unsigned short hb = f2bf(vals[i]);
      H.h[i] = hb;
      L.h[i] = f2bf(vals[i] - bf2f(hb));
    }
    // A k order used here (B staged the same, contiguous k): lanes 0-15 hold K {0..15}; lanes 16-31 {16..31}
    const int rmt = row >> 4, lane_lo = row & 15;
    const int kci = k >> 5,  kk = k & 31;
    const int lanehi = (kk >> 4) & 1;
    const int j0 = (kk & 8) ? 8 : 0;
    const int lane = lanehi * 16 + lane_lo;
    const int base = ((rmt * 32 + kci) * 32 + lane) * 16 + j0;
    *(v4u*)(aHi + base) = H.u;
    *(v4u*)(aLo + base) = L.u;
  }
  __syncthreads();

  float P = 0.f;
  float Z[NQ] = {0.f};

  for (int it = 0; it < 16; ++it) {
    const int n0 = it * 64;
    v8f accR = {};
    v8f accI = {};

    // ---- issue async stage of chunk 0 into buffer 0 (4 x B128 per thread) --
    {
#pragma unroll
      for (int ss = 0; ss < 2; ++ss) {
        const int s    = tid * 2 + ss;
        const int lane = s & 31;
        const int bsub = (s >> 5) & 3;
        const int part = (s >> 7) & 1;
        const int hl   = (s >> 8) & 1;
        const int col  = n0 + bsub * 16 + (lane & 15);
        const int kst  = ((lane >> 4) << 4);
        const unsigned short* src =
            (hl ? btlo : bthi) + (size_t)(part * DIM + col) * DIM + kst;
        async_copy_32B(bByte0 + (unsigned)s * 32u, src);
      }
    }

    for (int ks = 0; ks < 32; ++ks) {
      const int c = ks & 1;
      // ---- prefetch chunk ks+1 into the other buffer while ks computes ----
      if (ks + 1 < 32) {
#pragma unroll
        for (int ss = 0; ss < 2; ++ss) {
          const int s    = tid * 2 + ss;
          const int lane = s & 31;
          const int bsub = (s >> 5) & 3;
          const int part = (s >> 7) & 1;
          const int hl   = (s >> 8) & 1;
          const int col  = n0 + bsub * 16 + (lane & 15);
          const int kst  = (ks + 1) * 32 + ((lane >> 4) << 4);
          const unsigned short* src =
              (hl ? btlo : bthi) + (size_t)(part * DIM + col) * DIM + kst;
          async_copy_32B(bByte0 + (unsigned)(1 - c) * 16384u + (unsigned)s * 32u, src);
        }
        // async loads retire in order: <=4 outstanding means chunk ks landed
        asm volatile("s_wait_asynccnt 0x4" ::: "memory");
      } else {
        asm volatile("s_wait_asynccnt 0x0" ::: "memory");
      }
      __syncthreads();

      unsigned short* bStg = lds + 65536 + c * 8192;
      const int abase = ((mt * 32 + ks) * 32 + t) * 16;
      const v16bf ah = *(const v16bf*)(aHi + abase);
      const v16bf al = *(const v16bf*)(aLo + abase);
      const v16bf brh = *(const v16bf*)(bStg + (((0 * 2 + 0) * 4 + sub) * 32 + t) * 16);
      const v16bf bih = *(const v16bf*)(bStg + (((0 * 2 + 1) * 4 + sub) * 32 + t) * 16);
      const v16bf brl = *(const v16bf*)(bStg + (((1 * 2 + 0) * 4 + sub) * 32 + t) * 16);
      const v16bf bil = *(const v16bf*)(bStg + (((1 * 2 + 1) * 4 + sub) * 32 + t) * 16);

      // fp32-grade product via 3-term bf16 split: hi*hi + lo*hi + hi*lo
      accR = __builtin_amdgcn_wmma_f32_16x16x32_bf16(false, ah, false, brh, (short)0, accR, false, false);
      accR = __builtin_amdgcn_wmma_f32_16x16x32_bf16(false, al, false, brh, (short)0, accR, false, false);
      accR = __builtin_amdgcn_wmma_f32_16x16x32_bf16(false, ah, false, brl, (short)0, accR, false, false);
      accI = __builtin_amdgcn_wmma_f32_16x16x32_bf16(false, ah, false, bih, (short)0, accI, false, false);
      accI = __builtin_amdgcn_wmma_f32_16x16x32_bf16(false, al, false, bih, (short)0, accI, false, false);
      accI = __builtin_amdgcn_wmma_f32_16x16x32_bf16(false, ah, false, bil, (short)0, accI, false, false);
      __syncthreads();   // all waves done reading buffer c before it is refilled
    }

    // ---- |Y|^2 tile -> LDS (async writes drained; pbuf aliases buffer 0) ----
#pragma unroll
    for (int e = 0; e < 8; ++e) {
      const float pr = accR[e], pi = accI[e];
      const int rr = mt * 16 + e + ((t >> 4) << 3);   // C layout: lanes 16-31 hold M+8
      const int cc = sub * 16 + (t & 15);
      pbuf[rr * 64 + cc] = pr * pr + pi * pi;
    }
    __syncthreads();

    if (tid < TM) {
      for (int c = 0; c < 64; ++c) {
        const float p = pbuf[tid * 64 + c];
        const int kp = n0 + c;
        P += p;
#pragma unroll
        for (int q = 0; q < NQ; ++q)
          Z[q] += ((kp >> (9 - q)) & 1) ? p : 0.f;
      }
    }
    __syncthreads();   // epilogue reads done before next iteration's async issue
  }

  if (tid < TM) {
    const float inv = 1.0f / P;           // unitarity: P == ||x||^2, norm cancels
    const size_t rb = (size_t)(m0 + tid) * NQ;
#pragma unroll
    for (int q = 0; q < NQ; ++q)
      out[rb + q] = 1.0f - 2.0f * Z[q] * inv;
  }
}

// =====================================================================
extern "C" void kernel_launch(void* const* d_in, const int* in_sizes, int n_in,
                              void* d_out, int out_size, void* d_ws, size_t ws_size,
                              hipStream_t stream) {
  (void)n_in; (void)out_size; (void)ws_size;
  const float* x   = (const float*)d_in[0];
  const float* wts = (const float*)d_in[1];
  float* out = (float*)d_out;

  // workspace: Bt_hi [2048][1024] bf16 (4MB) | Bt_lo (4MB)
  unsigned short* bthi = (unsigned short*)d_ws;
  unsigned short* btlo = (unsigned short*)((char*)d_ws + (size_t)2 * DIM * DIM * sizeof(unsigned short));

  qsel_build_bt<<<DIM, THREADS, 0, stream>>>(wts, bthi, btlo);

  const int B = in_sizes[0] / DIM;   // 16384
  qsel_gemm<<<B / TM, THREADS, 0, stream>>>(x, bthi, btlo, out);
}
